// CasualSelfAttention_64304250355980
// MI455X (gfx1250) — matrix operand order repaired and checked by reading, not compiled, not run on hardware
//
#include <hip/hip_runtime.h>
#include <hip/hip_bf16.h>

// ---------------- Problem constants ----------------
#define BB   4
#define TT   2048
#define DD   1024
#define HH   16
#define DHD  64
#define M1   (BB * TT)          // 8192 token rows
#define N3D  (3 * DD)           // 3072
#define SCL  0.03125f           // D^-0.5 = 1/32

typedef __attribute__((ext_vector_type(16))) __bf16 v16bf;
typedef __attribute__((ext_vector_type(8)))  float  v8f;

union FragU { uint4 q[2]; v16bf v; };

// fp32 -> bf16 round-to-nearest-even
static __device__ __forceinline__ __bf16 f2bf(float f) {
    union { float f; unsigned int u; } a;
    a.f = f;
    unsigned int r = a.u + 0x7FFFu + ((a.u >> 16) & 1u);
    union { unsigned short s; __bf16 b; } o;
    o.s = (unsigned short)(r >> 16);
    return o.b;
}

// A-fragment (16x32 bf16, M=lane&15): lanes 0-15 hold K 0-7/16-23,
// lanes 16-31 hold K 8-15/24-31. Two 16B loads per lane.
static __device__ __forceinline__ v16bf frag_a(const __bf16* tile, int ldm) {
    int lane = threadIdx.x & 31;
    int r    = lane & 15;
    int k0   = (lane >> 4) << 3;
    FragU f;
    f.q[0] = *(const uint4*)(tile + r * ldm + k0);
    f.q[1] = *(const uint4*)(tile + r * ldm + k0 + 16);
    return f.v;
}

// B-fragment (32x16 bf16) from row-major [N][K] (i.e. B^T):
// N=lane&15; lanes 0-15 K 0-7/16-23, lanes 16-31 K 8-15/24-31. Two 16B loads.
static __device__ __forceinline__ v16bf frag_b(const __bf16* tileT, int ldm) {
    int lane = threadIdx.x & 31;
    int n    = lane & 15;
    int kb   = (lane >> 4) << 3;
    FragU f;
    f.q[0] = *(const uint4*)(tileT + n * ldm + kb);
    f.q[1] = *(const uint4*)(tileT + n * ldm + kb + 16);
    return f.v;
}

static __device__ __forceinline__ v8f wmma_bf16(v16bf a, v16bf b, v8f c) {
    return __builtin_amdgcn_wmma_f32_16x16x32_bf16(false, a, false, b,
                                                   (short)0, c, false, false);
}

static __device__ __forceinline__ v8f vzero() {
    return (v8f){0.f,0.f,0.f,0.f,0.f,0.f,0.f,0.f};
}

// =====================================================================
// Kernel 0a: elementwise fp32 -> bf16 (embds)
// =====================================================================
__global__ __launch_bounds__(256) void cvt_kernel(
        const float* __restrict__ src, __bf16* __restrict__ dst) {
    int i = (blockIdx.x * 256 + threadIdx.x) * 4;
    float4 x = *(const float4*)(src + i);
    union { __bf16 b[4]; uint2 u; } o;
    o.b[0] = f2bf(x.x); o.b[1] = f2bf(x.y);
    o.b[2] = f2bf(x.z); o.b[3] = f2bf(x.w);
    *(uint2*)(dst + i) = o.u;
}

// =====================================================================
// Kernel 0b: tiled transpose fp32[rows][cols] -> bf16[cols][rows]
// =====================================================================
__global__ __launch_bounds__(256) void transpose_kernel(
        const float* __restrict__ src, __bf16* __restrict__ dst,
        int rows, int cols) {
    __shared__ __bf16 t[32][33];
    const int tid = threadIdx.x;
    const int c0 = blockIdx.x * 32, r0 = blockIdx.y * 32;
#pragma unroll
    for (int i = 0; i < 4; ++i) {
        int e = tid + i * 256;
        int r = e >> 5, c = e & 31;
        t[r][c] = f2bf(src[(size_t)(r0 + r) * cols + c0 + c]);   // coalesced read
    }
    __syncthreads();
#pragma unroll
    for (int i = 0; i < 4; ++i) {
        int e = tid + i * 256;
        int r = e >> 5, c = e & 31;                               // r: col index
        dst[(size_t)(c0 + r) * rows + r0 + c] = t[c][r];          // coalesced write
    }
}

// =====================================================================
// Kernel 1: kqv = embds @ W_qkv (both bf16, B pre-transposed), scatter
// into head-split K/Q/V. Wave tile 32Mx64N (8 WMMA/kstep); block 64Mx256N.
// Pure global b128 fragment loads; no LDS, no barriers.
// =====================================================================
__global__ __launch_bounds__(256) void qkv_kernel(
        const __bf16* __restrict__ A, const __bf16* __restrict__ BT,
        __bf16* __restrict__ kws, __bf16* __restrict__ qws, __bf16* __restrict__ vws) {
    const int tid  = threadIdx.x;
    const int wave = tid >> 5, lane = tid & 31;
    const int wm = wave & 1;          // 2 waves in M -> 32 rows each
    const int wn = wave >> 1;         // 4 waves in N -> 64 cols each
    const int bm = blockIdx.x;        // 128 tiles of 64 rows
    const int bn = blockIdx.y;        // 12 tiles of 256 cols

    const __bf16* Ab = A  + (size_t)(bm * 64 + wm * 32) * DD;
    const __bf16* Bb = BT + (size_t)(bn * 256 + wn * 64) * DD;

    v8f acc[2][4];
#pragma unroll
    for (int h = 0; h < 2; ++h)
#pragma unroll
        for (int j = 0; j < 4; ++j) acc[h][j] = vzero();

    for (int k0 = 0; k0 < DD; k0 += 32) {
        __builtin_prefetch(Ab + k0 + 128, 0, 1);
        v16bf a0 = frag_a(Ab + k0, DD);
        v16bf a1 = frag_a(Ab + 16 * DD + k0, DD);
#pragma unroll
        for (int j = 0; j < 4; ++j) {
            v16bf b = frag_b(Bb + (size_t)(j * 16) * DD + k0, DD);
            acc[0][j] = wmma_bf16(a0, b, acc[0][j]);
            acc[1][j] = wmma_bf16(a1, b, acc[1][j]);
        }
    }

    const int hh = lane >> 4, nl = lane & 15;
#pragma unroll
    for (int half = 0; half < 2; ++half)
#pragma unroll
        for (int j = 0; j < 4; ++j)
#pragma unroll
            for (int g = 0; g < 8; ++g) {
                int m = bm * 64 + wm * 32 + half * 16 + g + 8 * hh;
                int n = bn * 256 + wn * 64 + j * 16 + nl;
                float v = acc[half][j][g];
                int b = m >> 11, t = m & (TT - 1);
                int sec = n >> 10, d = n & (DD - 1);
                int h = d >> 6, dh = d & 63;
                size_t idx = ((size_t)(b * HH + h) * TT + t) * DHD + dh;
                if (sec == 0)      kws[idx] = f2bf(v);
                else if (sec == 1) qws[idx] = f2bf(v * SCL);
                else               vws[idx] = f2bf(v);
            }
}

// =====================================================================
// Kernel 2: flash attention. One wave = one (b,h) x 16-query block.
// Per 32-key tile: 4 QK^T WMMAs + online softmax (max via shfl, SUM VIA
// WMMA against an all-ones B fragment) + 4 P*V WMMAs + 1 l-WMMA.
// Wave-private LDS only (divergent trip counts -> no block barriers).
// =====================================================================
__global__ __launch_bounds__(128) void attn_kernel(
        const __bf16* __restrict__ kws, const __bf16* __restrict__ qws,
        const __bf16* __restrict__ vws, __bf16* __restrict__ attn) {
    __shared__ __align__(16) __bf16 VT[4][64 * 40];  // V^T [dim][key] per wave
    __shared__ __align__(16) __bf16 PS[4][16 * 40];  // P   [q][key]   per wave

    const int tid  = threadIdx.x;
    const int w    = tid >> 5, lane = tid & 31;
    const int gw   = blockIdx.x * 4 + w;
    const int qb   = gw & 127;
    const int bh   = gw >> 7;

    const __bf16* Q = qws + (size_t)bh * TT * DHD;
    const __bf16* K = kws + (size_t)bh * TT * DHD;
    const __bf16* V = vws + (size_t)bh * TT * DHD;

    const v16bf aq0 = frag_a(Q + (size_t)qb * 16 * DHD, DHD);
    const v16bf aq1 = frag_a(Q + (size_t)qb * 16 * DHD + 32, DHD);

    // all-ones B fragment: C = P * ones -> per-row sums, replicated over lanes
    FragU of;
    of.q[0] = make_uint4(0x3F803F80u, 0x3F803F80u, 0x3F803F80u, 0x3F803F80u);
    of.q[1] = of.q[0];
    const v16bf ones = of.v;

    float mrow[8];
    v8f acc[4], accl;
#pragma unroll
    for (int g = 0; g < 8; ++g) mrow[g] = -3.0e38f;
#pragma unroll
    for (int n = 0; n < 4; ++n) acc[n] = vzero();
    accl = vzero();

    const int hh = lane >> 4, nl = lane & 15;
    const int ntiles = (qb + 2) >> 1;

    for (int kt = 0; kt < ntiles; ++kt) {
        const int kb = kt * 32;
        const __bf16* Kt = K + (size_t)kb * DHD;

        __builtin_prefetch(Kt + 32 * DHD, 0, 1);
        __builtin_prefetch(V + (size_t)(kb + 32 + lane) * DHD, 0, 1);

        // stage V^T: batch the 8 b128 loads, then scatter to LDS
        {
            const uint4* src = (const uint4*)(V + (size_t)(kb + lane) * DHD);
            uint4 vr[8];
#pragma unroll
            for (int c = 0; c < 8; ++c) vr[c] = src[c];
#pragma unroll
            for (int c = 0; c < 8; ++c) {
                const __bf16* xs = (const __bf16*)&vr[c];
#pragma unroll
                for (int e = 0; e < 8; ++e)
                    VT[w][(c * 8 + e) * 40 + lane] = xs[e];
            }
        }
        asm volatile("" ::: "memory");       // DS in-order within a wave

        // S = Q K^T
        v8f s0 = vzero(), s1 = vzero();
        s0 = wmma_bf16(aq0, frag_b(Kt, DHD), s0);
        s0 = wmma_bf16(aq1, frag_b(Kt + 32, DHD), s0);
        s1 = wmma_bf16(aq0, frag_b(Kt + 16 * DHD, DHD), s1);
        s1 = wmma_bf16(aq1, frag_b(Kt + 16 * DHD + 32, DHD), s1);

        // causal mask + online max; P to LDS; rescale accumulators
        const int key0 = kb + nl, key1 = kb + 16 + nl;
#pragma unroll
        for (int g = 0; g < 8; ++g) {
            int qrow = qb * 16 + g + 8 * hh;
            float v0 = (key0 <= qrow) ? s0[g] : -3.0e38f;
            float v1 = (key1 <= qrow) ? s1[g] : -3.0e38f;
            float mt = fmaxf(v0, v1);
#pragma unroll
            for (int off = 1; off < 16; off <<= 1)
                mt = fmaxf(mt, __shfl_xor(mt, off, 32));
            float mn = fmaxf(mrow[g], mt);
            float alpha = __expf(mrow[g] - mn);
            mrow[g] = mn;
            accl[g] *= alpha;
#pragma unroll
            for (int n = 0; n < 4; ++n) acc[n][g] *= alpha;
            PS[w][(g + 8 * hh) * 40 + nl]      = f2bf(__expf(v0 - mn));
            PS[w][(g + 8 * hh) * 40 + 16 + nl] = f2bf(__expf(v1 - mn));
        }
        asm volatile("" ::: "memory");

        // O += P V ; l += P 1
        v16bf ap = frag_a(PS[w], 40);
#pragma unroll
        for (int n = 0; n < 4; ++n) {
            v16bf bv = frag_b(&VT[w][(n * 16) * 40], 40);
            acc[n] = wmma_bf16(ap, bv, acc[n]);
        }
        accl = wmma_bf16(ap, ones, accl);
    }

    const int b = bh >> 4, h = bh & 15;
#pragma unroll
    for (int n = 0; n < 4; ++n)
#pragma unroll
        for (int g = 0; g < 8; ++g) {
            float o = acc[n][g] / accl[g];
            int t = qb * 16 + g + 8 * hh;
            size_t idx = ((size_t)(b * TT + t)) * DD + h * DHD + n * 16 + nl;
            attn[idx] = f2bf(o);
        }
}

// =====================================================================
// Kernel 3: out = attn @ W_proj + b_proj (fp32 out). Same pure-global
// WMMA GEMM; W_proj pre-transposed bf16.
// =====================================================================
__global__ __launch_bounds__(256) void proj_kernel(
        const __bf16* __restrict__ A, const __bf16* __restrict__ BT,
        const float* __restrict__ bias, float* __restrict__ out) {
    const int tid  = threadIdx.x;
    const int wave = tid >> 5, lane = tid & 31;
    const int wm = wave & 1;
    const int wn = wave >> 1;
    const int bm = blockIdx.x;        // 128 tiles of 64 rows
    const int bn = blockIdx.y;        // 4 tiles of 256 cols

    const __bf16* Ab = A  + (size_t)(bm * 64 + wm * 32) * DD;
    const __bf16* Bb = BT + (size_t)(bn * 256 + wn * 64) * DD;

    v8f acc[2][4];
#pragma unroll
    for (int h = 0; h < 2; ++h)
#pragma unroll
        for (int j = 0; j < 4; ++j) acc[h][j] = vzero();

    for (int k0 = 0; k0 < DD; k0 += 32) {
        __builtin_prefetch(Ab + k0 + 128, 0, 1);
        v16bf a0 = frag_a(Ab + k0, DD);
        v16bf a1 = frag_a(Ab + 16 * DD + k0, DD);
#pragma unroll
        for (int j = 0; j < 4; ++j) {
            v16bf b = frag_b(Bb + (size_t)(j * 16) * DD + k0, DD);
            acc[0][j] = wmma_bf16(a0, b, acc[0][j]);
            acc[1][j] = wmma_bf16(a1, b, acc[1][j]);
        }
    }

    const int hh = lane >> 4, nl = lane & 15;
#pragma unroll
    for (int half = 0; half < 2; ++half)
#pragma unroll
        for (int j = 0; j < 4; ++j)
#pragma unroll
            for (int g = 0; g < 8; ++g) {
                int m = bm * 64 + wm * 32 + half * 16 + g + 8 * hh;
                int n = bn * 256 + wn * 64 + j * 16 + nl;
                out[(size_t)m * DD + n] = acc[half][j][g] + bias[n];
            }
}

// =====================================================================
extern "C" void kernel_launch(void* const* d_in, const int* in_sizes, int n_in,
                              void* d_out, int out_size, void* d_ws, size_t ws_size,
                              hipStream_t stream) {
    const float* embds = (const float*)d_in[0];   // [B,T,D]
    const float* Wqkv  = (const float*)d_in[1];   // [D,3D]
    const float* Wproj = (const float*)d_in[2];   // [D,D]
    const float* bproj = (const float*)d_in[3];   // [D]

    const size_t eHead = (size_t)BB * HH * TT * DHD;  // 8,388,608
    char* ws = (char*)d_ws;
    __bf16* kws   = (__bf16*)(ws);
    __bf16* qws   = (__bf16*)(ws + eHead * 2);
    __bf16* vws   = (__bf16*)(ws + eHead * 4);
    __bf16* attn  = (__bf16*)(ws + eHead * 6);
    __bf16* ebf   = (__bf16*)(ws + eHead * 8);                         // [8192][1024]
    __bf16* WqkvT = (__bf16*)(ws + eHead * 10);                        // [3072][1024]
    __bf16* WprjT = (__bf16*)(ws + eHead * 10 + (size_t)N3D * DD * 2); // [1024][1024]

    cvt_kernel<<<dim3((M1 * DD) / 1024), 256, 0, stream>>>(embds, ebf);
    transpose_kernel<<<dim3(N3D / 32, DD / 32), 256, 0, stream>>>(Wqkv, WqkvT, DD, N3D);
    transpose_kernel<<<dim3(DD / 32, DD / 32), 256, 0, stream>>>(Wproj, WprjT, DD, DD);

    qkv_kernel<<<dim3(M1 / 64, N3D / 256), 256, 0, stream>>>(ebf, WqkvT, kws, qws, vws);
    attn_kernel<<<dim3((BB * HH * (TT / 16)) / 4), 128, 0, stream>>>(kws, qws, vws, attn);
    proj_kernel<<<dim3(M1 / 64, DD / 256), 256, 0, stream>>>(attn, WprjT, bproj, (float*)d_out);
}
